// WindowAttentionRPB_73040213835840
// MI455X (gfx1250) — matrix operand order repaired and checked by reading, not compiled, not run on hardware
//
#include <hip/hip_runtime.h>

// ---------------------------------------------------------------------------
// WindowAttentionRPB on MI455X (gfx1250): fully fused bf16 WMMA pipeline.
//   B=2048 windows, N=64 tokens, C=256, H=8 heads, d=32.
// One workgroup (8 wave32) per window. All intermediates (qkv, attn, attn@v)
// live in the 320KB/WGP LDS -> HBM traffic is just x-in + out + L2-hot weights.
// All matmuls: v_wmma_f32_16x16x32_bf16 with pre-swizzled register-image
// operands (ISA 7.12.2) so each lane does one 32B load per WMMA operand.
// 144KB LDS/block + <=256 VGPRs/wave => 2 workgroups per WGP.
// ---------------------------------------------------------------------------

typedef __attribute__((ext_vector_type(16))) __bf16 v16bf;
typedef __attribute__((ext_vector_type(8)))  float  v8f;

#define NWIN 2048   // number of windows (B_)
#define NTOK 64     // tokens per window
#define CDIM 256
#define HEADS 8
#define NW_MASK 64  // nW in mask

// LDS carve-up (bytes). xA region is reused as aA after stage 1.
#define XA_OFF 0        // 32KB: x in A-frag layout / later attn-out A-frags
#define QA_OFF 32768    // 32KB: q in A-frag layout   [h][rt] tiles
#define KB_OFF 65536    // 32KB: k^T in B-frag layout [h][ct] tiles
#define VB_OFF 98304    // 32KB: v in B-frag layout   [h][kt][nt] tiles
#define AT_OFF 131072   // 16KB: per-wave softmaxed-attn A-frag scratch
#define SMEM_BYTES 147456

// float -> bf16 (round to nearest even)
__device__ __forceinline__ unsigned short f2bf(float f) {
  union { float f; unsigned u; } v; v.f = f;
  unsigned u = v.u;
  return (unsigned short)((u + 0x7FFFu + ((u >> 16) & 1u)) >> 16);
}

// contiguous A-matrix element map (K = half*16 + e, matching the B fragments):
// given K in [0,32) find (element e, lane-half h)
__device__ __forceinline__ void kinv(int k, int& e, int& half) {
  half = (k >> 4) & 1;
  e = k & 15;
}

__device__ __forceinline__ v8f wmma_bf16(v16bf a, v16bf b, v8f c) {
  return __builtin_amdgcn_wmma_f32_16x16x32_bf16(false, a, false, b,
                                                 (short)0, c, false, false);
}

#define V8F_ZERO ((v8f){0.f, 0.f, 0.f, 0.f, 0.f, 0.f, 0.f, 0.f})

// ---------------------------------------------------------------------------
// Weight swizzle: W[out, 256] f32 -> bf16 B-fragment tiles (B = W^T).
// Tile index: ct*8 + kt ; element (lane,e): K = kt*32 + half*16 + e, N = ct*16+r.
// ---------------------------------------------------------------------------
__global__ void swz_w_kernel(const float* __restrict__ w,
                             unsigned short* __restrict__ wB, int ctCount) {
  int total = ctCount * 8 * 512;
  for (int idx = blockIdx.x * blockDim.x + threadIdx.x; idx < total;
       idx += gridDim.x * blockDim.x) {
    int e = idx & 15;
    int lane = (idx >> 4) & 31;
    int tile = idx >> 9;
    int kt = tile & 7;
    int ct = tile >> 3;
    int r = lane & 15, half = lane >> 4;
    int K = kt * 32 + half * 16 + e;
    int n = ct * 16 + r;
    wB[idx] = f2bf(w[n * CDIM + K]);
  }
}

// ---------------------------------------------------------------------------
// Fused per-window kernel: 256 threads = 8 waves, one window per block.
// amdgpu_waves_per_eu(4) caps VGPRs at 256/wave so two 8-wave workgroups
// (2 x 144KB LDS) co-reside per WGP.
// ---------------------------------------------------------------------------
__global__ void __launch_bounds__(256)
__attribute__((amdgpu_waves_per_eu(4)))
fused_window_attn(const float* __restrict__ x,
                  const unsigned short* __restrict__ qkvB,
                  const unsigned short* __restrict__ pB,
                  const float* __restrict__ qkv_b,
                  const float* __restrict__ proj_b,
                  const float* __restrict__ mask,
                  const float* __restrict__ rpb_table,
                  const int* __restrict__ rpb_index,
                  float* __restrict__ out) {
  extern __shared__ __attribute__((aligned(64))) char smem[];
  unsigned short* xAl = (unsigned short*)(smem + XA_OFF);
  unsigned short* qAl = (unsigned short*)(smem + QA_OFF);
  unsigned short* kBl = (unsigned short*)(smem + KB_OFF);
  unsigned short* vBl = (unsigned short*)(smem + VB_OFF);

  const long long b = blockIdx.x;
  const int tid = threadIdx.x;
  // wave id as a true SGPR so head / q-k-v selection become scalar branches
  const int wq = __builtin_amdgcn_readfirstlane(tid >> 5);
  const int lane = tid & 31;
  const int r = lane & 15, half = lane >> 4;

  // ---------------- Stage 0: x (f32, coalesced) -> LDS A-frag bf16 ----------
  // Consecutive channel pairs land in adjacent e-slots of the same fragment
  // row, so pack two bf16 into one b32 LDS store.
  {
    const float* xw = x + b * NTOK * CDIM;
    unsigned int* xAl32 = (unsigned int*)xAl;
#pragma unroll 4
    for (int i = 0; i < 16; ++i) {
      int fidx = i * 256 + tid;              // float4 index, 0..4095
      int n = fidx >> 6;                     // token row 0..63
      int c0 = (fidx & 63) << 2;             // starting channel (mult of 4)
      float4 val = ((const float4*)xw)[fidx];
      int rt = n >> 4;
      int rowbase = (n & 15) * 16;           // u16 offset of row within frag
      {
        int e2, h2; kinv(c0 & 31, e2, h2);
        unsigned int pk = (unsigned int)f2bf(val.x) |
                          ((unsigned int)f2bf(val.y) << 16);
        int u16i = (rt * 8 + (c0 >> 5)) * 512 + h2 * 256 + rowbase + e2;
        xAl32[u16i >> 1] = pk;
      }
      {
        int e2, h2; kinv((c0 + 2) & 31, e2, h2);
        unsigned int pk = (unsigned int)f2bf(val.z) |
                          ((unsigned int)f2bf(val.w) << 16);
        int u16i = (rt * 8 + (c0 >> 5)) * 512 + h2 * 256 + rowbase + e2;
        xAl32[u16i >> 1] = pk;
      }
    }
  }
  __syncthreads();

  // ---------------- Stage 1: QKV GEMM (M=64, N=768, K=256) ------------------
  // wave w handles column tiles ct = w, w+8, ..., w+40 (6 tiles) x 4 row tiles
  for (int ci = 0; ci < 6; ++ci) {
    int ct = wq + ci * 8;                 // scalar
    int which = ct >> 4;                  // scalar: 0=q 1=k 2=v
    int h = (ct & 15) >> 1;               // scalar head
    int dd = ((ct & 1) << 4) | r;         // per-lane dim within head
    const unsigned short* bBase = qkvB + ct * 8 * 512;
    if (ci < 5) __builtin_prefetch(bBase + 8 * 512, 0, 0);  // next ct weights
    float bval = qkv_b[ct * 16 + r];

    for (int rt = 0; rt < 4; ++rt) {
      v8f acc = V8F_ZERO;
#pragma unroll
      for (int kt = 0; kt < 8; ++kt) {
        v16bf a  = *(const v16bf*)(xAl + (rt * 8 + kt) * 512 + lane * 16);
        v16bf bb = *(const v16bf*)(bBase + kt * 512 + lane * 16);
        acc = wmma_bf16(a, bb, acc);
      }
      // scatter results (+bias) into attention-ready LDS layouts
      if (which == 0) {
        int e2, h2; kinv(dd, e2, h2);
#pragma unroll
        for (int v = 0; v < 8; ++v) {
          int n = rt * 16 + v + half * 8;
          qAl[(h * 4 + rt) * 512 + (h2 * 16 + (n & 15)) * 16 + e2] =
              f2bf(acc[v] + bval);
        }
      } else if (which == 1) {
        int lcol = ((dd >> 4) * 16) * 16 + (dd & 15);
#pragma unroll
        for (int v = 0; v < 8; ++v) {
          int n = rt * 16 + v + half * 8;
          kBl[(h * 4 + rt) * 512 + (n & 15) * 16 + lcol] = f2bf(acc[v] + bval);
        }
      } else {
#pragma unroll
        for (int v = 0; v < 8; ++v) {
          int n = rt * 16 + v + half * 8;
          vBl[((h * 2 + (n >> 5)) * 2 + (dd >> 4)) * 512 +
              (((n >> 4) & 1) * 16 + (dd & 15)) * 16 + (n & 15)] =
              f2bf(acc[v] + bval);
        }
      }
    }
  }
  __syncthreads();

  // ---------------- Stage 2: attention, wave = head -------------------------
  {
    int h = wq;                                             // scalar head
    unsigned short* aTl = (unsigned short*)(smem + AT_OFF) + wq * 1024;
    unsigned short* aAl = xAl;  // reuse stage-0 region for attn-out A-frags
    int mw = (int)(b & (NW_MASK - 1));
    const float scale = 0.17677669529663687f;  // 1/sqrt(32)

    for (int rt = 0; rt < 4; ++rt) {
      // ---- q @ k^T (K=32, one WMMA per column tile) ----
      v16bf aq = *(const v16bf*)(qAl + (h * 4 + rt) * 512 + lane * 16);
      float logit[4][8];
#pragma unroll
      for (int ct = 0; ct < 4; ++ct) {
        v16bf bk = *(const v16bf*)(kBl + (h * 4 + ct) * 512 + lane * 16);
        v8f acc = wmma_bf16(aq, bk, V8F_ZERO);
#pragma unroll
        for (int v = 0; v < 8; ++v) logit[ct][v] = acc[v];
      }
      // ---- * scale + rpb + mask ----
#pragma unroll
      for (int v = 0; v < 8; ++v) {
        int Mg = rt * 16 + v + half * 8;
#pragma unroll
        for (int ct = 0; ct < 4; ++ct) {
          int Ng = ct * 16 + r;
          int ridx = rpb_index[Mg * 64 + Ng];
          float add = rpb_table[ridx * HEADS + h] + mask[(mw * 64 + Mg) * 64 + Ng];
          logit[ct][v] = logit[ct][v] * scale + add;
        }
      }
      // ---- softmax per row; write probs pre-swizzled to A-frag layout ----
#pragma unroll
      for (int v = 0; v < 8; ++v) {
        float mx = fmaxf(fmaxf(logit[0][v], logit[1][v]),
                         fmaxf(logit[2][v], logit[3][v]));
#pragma unroll
        for (int m = 1; m <= 8; m <<= 1) mx = fmaxf(mx, __shfl_xor(mx, m, 32));
        float s = 0.f;
#pragma unroll
        for (int ct = 0; ct < 4; ++ct) {
          logit[ct][v] = __expf(logit[ct][v] - mx);
          s += logit[ct][v];
        }
#pragma unroll
        for (int m = 1; m <= 8; m <<= 1) s += __shfl_xor(s, m, 32);
        float inv = 1.f / s;
        int row = v + half * 8;
#pragma unroll
        for (int ct = 0; ct < 4; ++ct) {
          int col = ct * 16 + r;
          int e2, h2; kinv(col & 31, e2, h2);
          aTl[(col >> 5) * 512 + (h2 * 16 + row) * 16 + e2] =
              f2bf(logit[ct][v] * inv);
        }
      }
      // in-wave LDS transpose: make sure scattered stores land before loads
      asm volatile("s_wait_dscnt 0x0" ::: "memory");

      // ---- attn @ v (K=64 -> 2 WMMA steps, 2 output col tiles) ----
      v8f o0 = V8F_ZERO, o1 = V8F_ZERO;
#pragma unroll
      for (int kt2 = 0; kt2 < 2; ++kt2) {
        v16bf aa = *(const v16bf*)(aTl + kt2 * 512 + lane * 16);
        v16bf b0 = *(const v16bf*)(vBl + ((h * 2 + kt2) * 2 + 0) * 512 + lane * 16);
        v16bf b1 = *(const v16bf*)(vBl + ((h * 2 + kt2) * 2 + 1) * 512 + lane * 16);
        o0 = wmma_bf16(aa, b0, o0);
        o1 = wmma_bf16(aa, b1, o1);
      }
      // ---- store attn-out pre-swizzled A-layout: tile (rt, kstep=h) ----
      {
        int e2a, h2a, e2b, h2b;
        kinv(r, e2a, h2a);
        kinv(16 + r, e2b, h2b);
#pragma unroll
        for (int v = 0; v < 8; ++v) {
          int row = v + half * 8;
          aAl[(rt * 8 + h) * 512 + (h2a * 16 + row) * 16 + e2a] = f2bf(o0[v]);
          aAl[(rt * 8 + h) * 512 + (h2b * 16 + row) * 16 + e2b] = f2bf(o1[v]);
        }
      }
    }
  }
  __syncthreads();

  // ---------------- Stage 3: proj GEMM (M=64, N=256, K=256) -----------------
  for (int ci = 0; ci < 2; ++ci) {
    int ct = wq + ci * 8;                 // scalar
    const unsigned short* bBase = pB + ct * 8 * 512;
    float bval = proj_b[ct * 16 + r];
    for (int rt = 0; rt < 4; ++rt) {
      v8f acc = V8F_ZERO;
#pragma unroll
      for (int kt = 0; kt < 8; ++kt) {
        v16bf a  = *(const v16bf*)(xAl + (rt * 8 + kt) * 512 + lane * 16);
        v16bf bb = *(const v16bf*)(bBase + kt * 512 + lane * 16);
        acc = wmma_bf16(a, bb, acc);
      }
#pragma unroll
      for (int v = 0; v < 8; ++v) {
        int n = rt * 16 + v + half * 8;
        out[(b * NTOK + n) * CDIM + ct * 16 + r] = acc[v] + bval;
      }
    }
  }
}

// ---------------------------------------------------------------------------
extern "C" void kernel_launch(void* const* d_in, const int* in_sizes, int n_in,
                              void* d_out, int out_size, void* d_ws,
                              size_t ws_size, hipStream_t stream) {
  const float* x        = (const float*)d_in[0];
  const float* mask     = (const float*)d_in[1];
  const float* qkv_w    = (const float*)d_in[2];
  const float* qkv_b    = (const float*)d_in[3];
  const float* proj_w   = (const float*)d_in[4];
  const float* proj_b   = (const float*)d_in[5];
  const float* rpb_tab  = (const float*)d_in[6];
  const int*   rpb_idx  = (const int*)d_in[7];
  float* out = (float*)d_out;
  (void)in_sizes; (void)n_in; (void)out_size; (void)ws_size;

  // workspace: only the swizzled weights (~0.5 MB)
  char* ws = (char*)d_ws;
  unsigned short* qkvB = (unsigned short*)ws;                            // 384KB
  unsigned short* pB   = (unsigned short*)(ws + 48 * 8 * 512 * 2 + 256); // 128KB

  swz_w_kernel<<<384, 256, 0, stream>>>(qkv_w, qkvB, 48);
  swz_w_kernel<<<128, 256, 0, stream>>>(proj_w, pB, 16);

  // allow >64KB dynamic LDS (320KB/WGP on CDNA5)
  hipFuncSetAttribute((const void*)fused_window_attn,
                      hipFuncAttributeMaxDynamicSharedMemorySize, SMEM_BYTES);

  fused_window_attn<<<NWIN, 256, SMEM_BYTES, stream>>>(
      x, qkvB, pB, qkv_b, proj_b, mask, rpb_tab, rpb_idx, out);
}
